// SPADE_33981781246328
// MI455X (gfx1250) — matrix operand order repaired and checked by reading, not compiled, not run on hardware
//
#include <hip/hip_runtime.h>
#include <math.h>

// ---------------- shapes (fixed by the reference) ----------------
#define BN   2
#define HH   96
#define WWX  96
#define HWX  (HH * WWX)          // 9216
#define NCX  64
#define SCX  64
#define MIDX 16
#define PTILES ((BN * HWX) / 16) // 1152 pixel tiles (16 contiguous along W)
#define TPB_ROW (WWX / 16)       // 6 tiles per image row
#define TPB_IMG (HH * TPB_ROW)   // 576 tiles per batch image

typedef __attribute__((ext_vector_type(16))) _Float16 v16h;
typedef __attribute__((ext_vector_type(8)))  _Float16 v8h;
typedef __attribute__((ext_vector_type(8)))  float    v8f;
typedef __attribute__((ext_vector_type(4)))  int      v4i;

// ---- cooperative stage of a weight-fragment block into LDS ----
// Uses the CDNA5 async global->LDS path when the toolchain exposes it
// (ASYNCcnt-tracked, bypasses VGPRs); falls back to a register copy.
__device__ __forceinline__ void stage_to_lds(const _Float16* __restrict__ g,
                                             _Float16* l, int halves) {
#if __has_builtin(__builtin_amdgcn_global_load_async_to_lds_b128)
  typedef __attribute__((address_space(1))) v4i* gp_t;
  typedef __attribute__((address_space(3))) v4i* lp_t;
  for (int i = (int)threadIdx.x * 8; i < halves; i += (int)blockDim.x * 8) {
    __builtin_amdgcn_global_load_async_to_lds_b128((gp_t)(g + i), (lp_t)(l + i), 0, 0);
  }
#if __has_builtin(__builtin_amdgcn_s_wait_asynccnt)
  __builtin_amdgcn_s_wait_asynccnt(0);
#else
  asm volatile("s_wait_asynccnt 0x0" ::: "memory");
#endif
#else
  for (int i = (int)threadIdx.x * 8; i < halves; i += (int)blockDim.x * 8) {
    *(v8h*)(l + i) = *(const v8h*)(g + i);
  }
#endif
  __syncthreads();
}

// =======================================================================
// Pack kernels: build A and B matrices in EXACT WMMA fragment order so the
// GEMM loops are pure {contiguous b128 loads + v_wmma}.
//
// A fragment order (16x32 f16): lane<16 holds K {0..7,16..23}, lane>=16
// holds {8..15,24..31}; row M = lane&15.  Flat: ((pt*nkb+kbi)*32+lane)*16+e.
// B fragment order (32x16 f16): lane<16 K {0..7,16..23}, lane>=16 K {8..15,24..31},
// col N = lane&15.  Flat: ((kbi*ntc+nt)*32+lane)*16+e.
// =======================================================================

__global__ void spade_pack_wfrag(const float* __restrict__ w, _Float16* __restrict__ wf,
                                 int Cout, int Cin, int nkb, int ntc) {
  int t = blockIdx.x * blockDim.x + threadIdx.x;
  int total = nkb * ntc * 512;
  if (t >= total) return;
  int e = t & 15;
  int lane = (t >> 4) & 31;
  int rest = t >> 9;
  int nt = rest % ntc, kbi = rest / ntc;
  int K = Cin * 9;
  int k = kbi * 32 + ((e < 8) ? e : e + 8) + ((lane < 16) ? 0 : 8);
  int oc = nt * 16 + (lane & 15);
  float v = (k < K && oc < Cout) ? w[oc * K + k] : 0.0f;
  wf[t] = (_Float16)v;
}

__global__ void spade_pack_a_im2col(const float* __restrict__ src0, int C0,
                                    const float* __restrict__ src1, int C1,
                                    _Float16* __restrict__ apack, int nkb) {
  int t = blockIdx.x * blockDim.x + threadIdx.x;
  int total = PTILES * nkb * 512;
  if (t >= total) return;
  int e = t & 15;
  int lane = (t >> 4) & 31;
  int rest = t >> 9;
  int kbi = rest % nkb, pt = rest / nkb;
  int b = pt / TPB_IMG, r = pt % TPB_IMG;
  int h = r / TPB_ROW;
  int wc = (r % TPB_ROW) * 16 + (lane & 15);
  int ka = kbi * 32 + ((e < 8) ? e : e + 8) + ((lane < 16) ? 0 : 8);
  int K = (C0 + C1) * 9;
  float v = 0.0f;
  if (ka < K) {
    int cin = ka / 9, tap = ka - cin * 9;
    int hh2 = h + tap / 3 - 1;
    int ww2 = wc + tap % 3 - 1;
    if (hh2 >= 0 && hh2 < HH && ww2 >= 0 && ww2 < WWX) {
      v = (cin < C0)
            ? src0[((size_t)(b * C0 + cin)) * HWX + hh2 * WWX + ww2]
            : src1[((size_t)(b * C1 + (cin - C0))) * HWX + hh2 * WWX + ww2];
    }
  }
  apack[t] = (_Float16)v;
}

// ---- DCNv2 sampled activations (mask * bilinear) -> A-fragment f16 ----
__global__ void spade_pack_a_dcn(const float* __restrict__ feat,
                                 const float* __restrict__ pyb,
                                 const float* __restrict__ pxb,
                                 const float* __restrict__ mkb,
                                 _Float16* __restrict__ apack, int nkb) {
  int t = blockIdx.x * blockDim.x + threadIdx.x;
  int total = PTILES * nkb * 512;
  if (t >= total) return;
  int e = t & 15;
  int lane = (t >> 4) & 31;
  int rest = t >> 9;
  int kbi = rest % nkb, pt = rest / nkb;
  int b = pt / TPB_IMG, r = pt % TPB_IMG;
  int h = r / TPB_ROW;
  int wc = (r % TPB_ROW) * 16 + (lane & 15);
  int ka = kbi * 32 + ((e < 8) ? e : e + 8) + ((lane < 16) ? 0 : 8);
  int c = ka / 9, kk = ka - c * 9;
  int pix = h * WWX + wc;
  int sidx = (b * 9 + kk) * HWX + pix;
  float py = pyb[sidx], px = pxb[sidx], mk = mkb[sidx];
  float fy = floorf(py), fx = floorf(px);
  int y0 = (int)fy, x0 = (int)fx;
  float wy = py - fy, wx = px - fx;
  const float* fc = feat + ((size_t)(b * SCX + c)) * HWX;
  float v = 0.0f;
  bool yi0 = (y0 >= 0) && (y0 < HH);
  bool yi1 = (y0 + 1 >= 0) && (y0 + 1 < HH);
  bool xi0 = (x0 >= 0) && (x0 < WWX);
  bool xi1 = (x0 + 1 >= 0) && (x0 + 1 < WWX);
  if (yi0 && xi0) v += (1.f - wy) * (1.f - wx) * fc[y0 * WWX + x0];
  if (yi0 && xi1) v += (1.f - wy) * wx         * fc[y0 * WWX + x0 + 1];
  if (yi1 && xi0) v += wy * (1.f - wx)         * fc[(y0 + 1) * WWX + x0];
  if (yi1 && xi1) v += wy * wx                 * fc[(y0 + 1) * WWX + x0 + 1];
  apack[t] = (_Float16)(mk * v);
}

// ---------------- positional norm across C (ddof=1) ----------------
__global__ void spade_posnorm(const float* __restrict__ x, float* __restrict__ nrm) {
  int p = blockIdx.x * blockDim.x + threadIdx.x;
  if (p >= BN * HWX) return;
  int b = p / HWX, off = p % HWX;
  const float* base = x + (size_t)b * NCX * HWX + off;
  float s = 0.f, s2 = 0.f;
  for (int c = 0; c < NCX; ++c) {
    float v = base[c * HWX];
    s += v; s2 += v * v;
  }
  float mean = s * (1.0f / NCX);
  float var = (s2 - NCX * mean * mean) * (1.0f / (NCX - 1));
  float inv = rsqrtf(var + 1e-5f);
  float* dst = nrm + (size_t)b * NCX * HWX + off;
  for (int c = 0; c < NCX; ++c)
    dst[c * HWX] = (base[c * HWX] - mean) * inv;
}

// ---------------- offset/mask precompute from om (B,27,H,W) ----------------
__global__ void spade_offparams(const float* __restrict__ om,
                                float* __restrict__ pyb, float* __restrict__ pxb,
                                float* __restrict__ mkb) {
  int t = blockIdx.x * blockDim.x + threadIdx.x;
  if (t >= BN * 9 * HWX) return;
  int b = t / (9 * HWX);
  int r = t % (9 * HWX);
  int kk = r / HWX, off = r % HWX;
  int h = off / WWX, w = off % WWX;
  const float* omb = om + (size_t)b * 27 * HWX;
  float dy = omb[(2 * kk) * HWX + off];
  float dx = omb[(2 * kk + 1) * HWX + off];
  float mv = omb[(18 + kk) * HWX + off];
  float mk = 1.0f / (1.0f + expf(-mv));
  int idx = (b * 9 + kk) * HWX + off;
  pyb[idx] = dy + (float)h + (float)(kk / 3 - 1);
  pxb[idx] = dx + (float)w + (float)(kk % 3 - 1);
  mkb[idx] = mk;
}

// =======================================================================
// GEMM: one wave owns 16 pixels x (NT*16) out-channels.  Weights are staged
// once per block into LDS (shared by all 8 waves); the K loop is then
// {1 global A load + prefetch + NT ds_load_b128 + NT v_wmma}.
// =======================================================================
template <int NT, bool RELU>
__global__ void __launch_bounds__(256)
spade_gemm_wmma(const _Float16* __restrict__ apack,
                const _Float16* __restrict__ wfrag,
                const float* __restrict__ bias,
                float* __restrict__ dst, int Cout, int nkb) {
  extern __shared__ _Float16 smem[];
  stage_to_lds(wfrag, smem, nkb * NT * 512);

  int gw = (blockIdx.x * blockDim.x + threadIdx.x) >> 5;
  int lane = threadIdx.x & 31;
  if (gw >= PTILES) return;                 // wave-uniform: EXEC all-1s below
  int b = gw / TPB_IMG, r = gw % TPB_IMG;
  int h = r / TPB_ROW, wb = (r % TPB_ROW) * 16;
  int m = lane & 15;
  int lh = (lane < 16) ? 0 : 1;
  const _Float16* ap = apack + ((size_t)gw * nkb) * 512 + lane * 16;
  const _Float16* wl = smem + (size_t)lane * 16;

  v8f acc[NT] = {};
  for (int kbi = 0; kbi < nkb; ++kbi) {
    v16h a = *(const v16h*)(ap + (size_t)kbi * 512);
    __builtin_prefetch(ap + (size_t)(kbi + 1) * 512, 0, 0);
#pragma unroll
    for (int nt = 0; nt < NT; ++nt) {
      v16h bf = *(const v16h*)(wl + (size_t)(kbi * NT + nt) * 512);
      acc[nt] = __builtin_amdgcn_wmma_f32_16x16x32_f16(false, a, false, bf,
                                                       (short)0, acc[nt], false, false);
    }
  }
#pragma unroll
  for (int nt = 0; nt < NT; ++nt) {
    int oc = nt * 16 + m;
    if (oc < Cout) {
      float bb = bias[oc];
#pragma unroll
      for (int rr = 0; rr < 8; ++rr) {
        int mm = rr + lh * 8;               // D: M = rr + 8*(lane>=16)
        float v = acc[nt][rr] + bb;
        if (RELU) v = fmaxf(v, 0.0f);
        dst[((size_t)(b * Cout + oc)) * HWX + h * WWX + (wb + mm)] = v;
      }
    }
  }
}

// ------- fused tail: gamma/beta GEMMs (shared A frags) + SPADE combine -----
__global__ void __launch_bounds__(256)
spade_tail_wmma(const _Float16* __restrict__ apack,   // packed from ref_refine
                const _Float16* __restrict__ wfg, const _Float16* __restrict__ wfb,
                const float* __restrict__ bg, const float* __restrict__ bbe,
                const float* __restrict__ nrm, float* __restrict__ out, int nkb) {
  const int NT = NCX / 16;                  // 4
  extern __shared__ _Float16 smem[];
  int sz = nkb * NT * 512;
  stage_to_lds(wfg, smem, sz);
  stage_to_lds(wfb, smem + sz, sz);

  int gw = (blockIdx.x * blockDim.x + threadIdx.x) >> 5;
  int lane = threadIdx.x & 31;
  if (gw >= PTILES) return;
  int b = gw / TPB_IMG, r = gw % TPB_IMG;
  int h = r / TPB_ROW, wb = (r % TPB_ROW) * 16;
  int m = lane & 15;
  int lh = (lane < 16) ? 0 : 1;
  const _Float16* ap = apack + ((size_t)gw * nkb) * 512 + lane * 16;
  const _Float16* wg = smem + (size_t)lane * 16;
  const _Float16* wbp = smem + sz + (size_t)lane * 16;

  v8f accg[NT] = {}, accb[NT] = {};
  for (int kbi = 0; kbi < nkb; ++kbi) {
    v16h a = *(const v16h*)(ap + (size_t)kbi * 512);
    __builtin_prefetch(ap + (size_t)(kbi + 1) * 512, 0, 0);
#pragma unroll
    for (int nt = 0; nt < NT; ++nt) {
      v16h bG = *(const v16h*)(wg + (size_t)(kbi * NT + nt) * 512);
      v16h bB = *(const v16h*)(wbp + (size_t)(kbi * NT + nt) * 512);
      accg[nt] = __builtin_amdgcn_wmma_f32_16x16x32_f16(false, a, false, bG,
                                                        (short)0, accg[nt], false, false);
      accb[nt] = __builtin_amdgcn_wmma_f32_16x16x32_f16(false, a, false, bB,
                                                        (short)0, accb[nt], false, false);
    }
  }
#pragma unroll
  for (int nt = 0; nt < NT; ++nt) {
    int oc = nt * 16 + m;
    float g0 = bg[oc], be0 = bbe[oc];
#pragma unroll
    for (int rr = 0; rr < 8; ++rr) {
      int mm = rr + lh * 8;
      size_t idx = ((size_t)(b * NCX + oc)) * HWX + h * WWX + (wb + mm);
      out[idx] = nrm[idx] * (1.0f + (accg[nt][rr] + g0)) + (accb[nt][rr] + be0);
    }
  }
}

// ------------------------------- launcher -------------------------------
extern "C" void kernel_launch(void* const* d_in, const int* in_sizes, int n_in,
                              void* d_out, int out_size, void* d_ws, size_t ws_size,
                              hipStream_t stream) {
  const float* x     = (const float*)d_in[0];
  const float* ref   = (const float*)d_in[1];
  const float* w1    = (const float*)d_in[2];
  const float* b1    = (const float*)d_in[3];
  const float* w2    = (const float*)d_in[4];
  const float* b2    = (const float*)d_in[5];
  const float* w_off = (const float*)d_in[6];
  const float* b_off = (const float*)d_in[7];
  const float* w_dcn = (const float*)d_in[8];
  const float* b_dcn = (const float*)d_in[9];
  const float* w_g   = (const float*)d_in[10];
  const float* b_g   = (const float*)d_in[11];
  const float* w_be  = (const float*)d_in[12];
  const float* b_be  = (const float*)d_in[13];
  float* out = (float*)d_out;

  // workspace carve-out; 64B-align every buffer so v16h loads are b128-clean
  float* ws = (float*)d_ws;
  size_t off = 0;
  auto falloc = [&](size_t n) {
    float* p = ws + off; off += (n + 15) & ~(size_t)15; return p;
  };
  auto halloc = [&](size_t nh) {
    _Float16* p = (_Float16*)(ws + off);
    off += (((nh + 1) / 2) + 15) & ~(size_t)15; return p;
  };

  float* nrm   = falloc((size_t)BN * NCX * HWX);
  float* hbuf  = falloc((size_t)BN * MIDX * HWX);
  float* cond  = falloc((size_t)BN * SCX * HWX);
  float* om    = falloc((size_t)BN * 27 * HWX);
  float* pyb   = falloc((size_t)BN * 9 * HWX);
  float* pxb   = falloc((size_t)BN * 9 * HWX);
  float* mkb   = falloc((size_t)BN * 9 * HWX);
  float* refrf = falloc((size_t)BN * SCX * HWX);

  // K-block counts (K padded to multiple of 32)
  const int nkb1 = 36;  // conv1: K=1152
  const int nkb2 = 5;   // conv2: K=144 -> 160
  const int nkbo = 18;  // offset conv: K=576
  const int nkbd = 18;  // dcn / gamma / beta: K=576

  // B-fragment buffers
  _Float16* wf1  = halloc((size_t)nkb1 * 1 * 512);
  _Float16* wf2  = halloc((size_t)nkb2 * 4 * 512);
  _Float16* wfo  = halloc((size_t)nkbo * 2 * 512);
  _Float16* wfd  = halloc((size_t)nkbd * 4 * 512);
  _Float16* wfg  = halloc((size_t)nkbd * 4 * 512);
  _Float16* wfbe = halloc((size_t)nkbd * 4 * 512);

  // one reusable A-fragment buffer, sized for the largest GEMM (conv1)
  _Float16* apack = halloc((size_t)PTILES * nkb1 * 512);

  auto cdiv = [](size_t a, size_t b) { return (int)((a + b - 1) / b); };

  // 1) pack all weights into B-fragment order
  spade_pack_wfrag<<<cdiv((size_t)nkb1 * 1 * 512, 256), 256, 0, stream>>>(w1, wf1, MIDX, NCX + SCX, nkb1, 1);
  spade_pack_wfrag<<<cdiv((size_t)nkb2 * 4 * 512, 256), 256, 0, stream>>>(w2, wf2, SCX, MIDX, nkb2, 4);
  spade_pack_wfrag<<<cdiv((size_t)nkbo * 2 * 512, 256), 256, 0, stream>>>(w_off, wfo, 27, SCX, nkbo, 2);
  spade_pack_wfrag<<<cdiv((size_t)nkbd * 4 * 512, 256), 256, 0, stream>>>(w_dcn, wfd, SCX, SCX, nkbd, 4);
  spade_pack_wfrag<<<cdiv((size_t)nkbd * 4 * 512, 256), 256, 0, stream>>>(w_g, wfg, NCX, SCX, nkbd, 4);
  spade_pack_wfrag<<<cdiv((size_t)nkbd * 4 * 512, 256), 256, 0, stream>>>(w_be, wfbe, NCX, SCX, nkbd, 4);

  // 2) positional norm
  spade_posnorm<<<cdiv(BN * HWX, 256), 256, 0, stream>>>(x, nrm);

  const int gemm_blocks = cdiv((size_t)PTILES * 32, 256);

  // 3) conv1: concat(nrm, ref) -> h (relu), Cout=16
  spade_pack_a_im2col<<<cdiv((size_t)PTILES * nkb1 * 512, 256), 256, 0, stream>>>(nrm, NCX, ref, SCX, apack, nkb1);
  spade_gemm_wmma<1, true><<<gemm_blocks, 256, (size_t)nkb1 * 1 * 512 * 2, stream>>>(apack, wf1, b1, hbuf, MIDX, nkb1);

  // 4) conv2: h -> cond (relu), Cout=64
  spade_pack_a_im2col<<<cdiv((size_t)PTILES * nkb2 * 512, 256), 256, 0, stream>>>(hbuf, MIDX, nullptr, 0, apack, nkb2);
  spade_gemm_wmma<4, true><<<gemm_blocks, 256, (size_t)nkb2 * 4 * 512 * 2, stream>>>(apack, wf2, b2, cond, SCX, nkb2);

  // 5) offset conv: cond -> om, Cout=27
  spade_pack_a_im2col<<<cdiv((size_t)PTILES * nkbo * 512, 256), 256, 0, stream>>>(cond, SCX, nullptr, 0, apack, nkbo);
  spade_gemm_wmma<2, false><<<gemm_blocks, 256, (size_t)nkbo * 2 * 512 * 2, stream>>>(apack, wfo, b_off, om, 27, nkbo);

  // 6) sampling params (py, px, sigmoid mask)
  spade_offparams<<<cdiv(BN * 9 * HWX, 256), 256, 0, stream>>>(om, pyb, pxb, mkb);

  // 7) deformable conv einsum -> ref_refine
  spade_pack_a_dcn<<<cdiv((size_t)PTILES * nkbd * 512, 256), 256, 0, stream>>>(ref, pyb, pxb, mkb, apack, nkbd);
  spade_gemm_wmma<4, false><<<gemm_blocks, 256, (size_t)nkbd * 4 * 512 * 2, stream>>>(apack, wfd, b_dcn, refrf, SCX, nkbd);

  // 8) fused gamma/beta GEMMs + SPADE combine -> out
  spade_pack_a_im2col<<<cdiv((size_t)PTILES * nkbd * 512, 256), 256, 0, stream>>>(refrf, SCX, nullptr, 0, apack, nkbd);
  spade_tail_wmma<<<gemm_blocks, 256, (size_t)nkbd * 4 * 512 * 2 * 2, stream>>>(apack, wfg, wfbe, b_g, b_be, nrm, out, nkbd);

  (void)in_sizes; (void)n_in; (void)out_size; (void)ws_size;
}
